// DirectVoxGO_33578054320534
// MI455X (gfx1250) — matrix operand order repaired and checked by reading, not compiled, not run on hardware
//
#include <hip/hip_runtime.h>
#include <math.h>

#define GRIDN 256
#define G3 (GRIDN * GRIDN * GRIDN)
#define NS 256

typedef __attribute__((ext_vector_type(16))) _Float16 v16h;
typedef __attribute__((ext_vector_type(8)))  _Float16 v8h;
typedef __attribute__((ext_vector_type(8)))  float    v8f;

union V16 { v16h v; v8h h[2]; };

__device__ __forceinline__ v8f wmma_f16(v16h a, v16h b, v8f c) {
    // D = A(16x32 f16) * B(32x16 f16) + C(16x16 f32)
    return __builtin_amdgcn_wmma_f32_16x16x32_f16(
        /*neg_a=*/false, a, /*neg_b=*/false, b,
        /*c_mod=*/(short)0, c, /*reuse_a=*/false, /*reuse_b=*/false);
}

__device__ __forceinline__ float softplus_f(float x) {
    return (x > 15.f) ? x : log1pf(__expf(x));
}

__global__ __launch_bounds__(256)
void dvgo_fused(const float* __restrict__ rays_pts,
                const float* __restrict__ viewdirs,
                const float* __restrict__ density,
                const float* __restrict__ k0,
                const float* __restrict__ w1,
                const float* __restrict__ b1,
                const float* __restrict__ w2,
                const float* __restrict__ b2,
                float* __restrict__ out)
{
    // LDS: padded feature rows (A-matrix staging), per-wave hidden tiles,
    // K-major transposed weights (B-matrix staging), scan + weights + accum.
    __shared__ _Float16 P[NS][64];        // [sample][K pad 64]  : 32 KB
    __shared__ _Float16 Hs[8][16][32];    // per-wave hidden tile:  8 KB
    __shared__ _Float16 W1T[16][64];      // w1^T, K padded to 64:  2 KB
    __shared__ _Float16 W2T[16][32];      // w2^T, K padded to 32:  1 KB
    __shared__ float scanLds[NS];
    __shared__ float wLds[NS];
    __shared__ float vembLds[27];
    __shared__ float accLds[3];
    __shared__ float bgLds;

    const int ray = blockIdx.x;
    const int s   = threadIdx.x;          // one sample per thread

    if (s < 3) accLds[s] = 0.f;

    // ---------------- trilinear sampling (density + 7-ch k0) ----------------
    const float* pp = rays_pts + ((size_t)ray * NS + s) * 3;
    float gx = (pp[0] + 1.f) * 0.5f * (float)(GRIDN - 1);
    float gy = (pp[1] + 1.f) * 0.5f * (float)(GRIDN - 1);
    float gz = (pp[2] + 1.f) * 0.5f * (float)(GRIDN - 1);
    gx = fminf(fmaxf(gx, 0.f), (float)(GRIDN - 1));
    gy = fminf(fmaxf(gy, 0.f), (float)(GRIDN - 1));
    gz = fminf(fmaxf(gz, 0.f), (float)(GRIDN - 1));
    int ix = min((int)floorf(gx), GRIDN - 2);
    int iy = min((int)floorf(gy), GRIDN - 2);
    int iz = min((int)floorf(gz), GRIDN - 2);
    float fx = gx - (float)ix, fy = gy - (float)iy, fz = gz - (float)iz;
    float wx0 = 1.f - fx, wy0 = 1.f - fy, wz0 = 1.f - fz;

    const int dX = GRIDN * GRIDN, dY = GRIDN;
    const int ibase = (ix * GRIDN + iy) * GRIDN + iz;
    const float c000 = wx0 * wy0 * wz0, c001 = wx0 * wy0 * fz;
    const float c010 = wx0 * fy  * wz0, c011 = wx0 * fy  * fz;
    const float c100 = fx  * wy0 * wz0, c101 = fx  * wy0 * fz;
    const float c110 = fx  * fy  * wz0, c111 = fx  * fy  * fz;

    float sigma =
        c000 * density[ibase]           + c001 * density[ibase + 1] +
        c010 * density[ibase + dY]      + c011 * density[ibase + dY + 1] +
        c100 * density[ibase + dX]      + c101 * density[ibase + dX + 1] +
        c110 * density[ibase + dX + dY] + c111 * density[ibase + dX + dY + 1];

    const float ACT_SHIFT = -4.595119850134589f;   // log(1/(1-0.01) - 1)
    float alpha = 1.f - __expf(-softplus_f(sigma + ACT_SHIFT));

    float feat[7];
    #pragma unroll
    for (int c = 0; c < 7; ++c) {
        const float* g = k0 + (size_t)c * (size_t)G3;
        feat[c] =
            c000 * g[ibase]           + c001 * g[ibase + 1] +
            c010 * g[ibase + dY]      + c011 * g[ibase + dY + 1] +
            c100 * g[ibase + dX]      + c101 * g[ibase + dX + 1] +
            c110 * g[ibase + dX + dY] + c111 * g[ibase + dX + dY + 1];
    }

    // ---------------- stage features / weights / view embedding ----------------
    #pragma unroll
    for (int c = 0; c < 7; ++c) P[s][c] = (_Float16)feat[c];
    #pragma unroll
    for (int j = 34; j < 64; ++j) P[s][j] = (_Float16)0.f;

    float la = __logf(fmaxf(1.f - alpha, 1e-10f));
    scanLds[s] = la;

    if (s < 3) {                                   // view embedding (per ray)
        float v = viewdirs[(size_t)ray * 3 + s];
        vembLds[s] = v;
        #pragma unroll
        for (int f = 0; f < 4; ++f) {
            float a = v * (float)(1 << f);
            vembLds[3  + s * 4 + f] = __sinf(a);
            vembLds[15 + s * 4 + f] = __cosf(a);
        }
    }
    for (int i = s; i < 16 * 64; i += 256) {       // w1^T K-major, zero-padded
        int n = i >> 6, k = i & 63;
        W1T[n][k] = (k < 34) ? (_Float16)w1[k * 16 + n] : (_Float16)0.f;
    }
    for (int i = s; i < 16 * 32; i += 256) {       // w2^T K-major, zero-padded
        int n = i >> 5, k = i & 31;
        W2T[n][k] = (k < 16 && n < 3) ? (_Float16)w2[k * 3 + n] : (_Float16)0.f;
    }
    __syncthreads();

    #pragma unroll
    for (int j = 0; j < 27; ++j) P[s][7 + j] = (_Float16)vembLds[j];

    // ---------------- cumprod via log-space inclusive scan ----------------
    for (int off = 1; off < NS; off <<= 1) {
        float a = scanLds[s];
        float b = (s >= off) ? scanLds[s - off] : 0.f;
        __syncthreads();
        scanLds[s] = a + b;
        __syncthreads();
    }
    float incl = scanLds[s];
    float weight = alpha * __expf(incl - la);      // alpha * exclusive cumprod
    wLds[s] = (weight > 1e-4f) ? weight : 0.f;     // FAST_COLOR_THRES
    if (s == NS - 1) bgLds = __expf(incl);         // alphainv_cum[-1]
    __syncthreads();

    // ---------------- MLP on 16-sample tiles via WMMA ----------------
    const int wave = s >> 5;
    const int lane = s & 31;
    const int nIdx = lane & 15;                    // row (A) / col (B,C,D)
    const int hi   = lane >> 4;                    // lane half selects K block

    // zero the padded K=16..31 region of this wave's hidden tile
    {
        v8h z8 = {(_Float16)0, (_Float16)0, (_Float16)0, (_Float16)0,
                  (_Float16)0, (_Float16)0, (_Float16)0, (_Float16)0};
        ((v8h*)Hs[wave][nIdx])[2 + hi] = z8;
    }
    float b1n = b1[nIdx];
    float b2n = (nIdx < 3) ? b2[nIdx] : 0.f;

    #pragma unroll
    for (int t = 0; t < 2; ++t) {
        const int tile = wave * 2 + t;
        const int s0   = tile * 16;

        // A fragments: lane l holds row (l&15); K chunks {kb..kb+7, 16+kb..}
        const v8h* prow = (const v8h*)P[s0 + nIdx];
        V16 aLo, aHi;
        aLo.h[0] = prow[hi];     aLo.h[1] = prow[2 + hi];   // K 0..31
        aHi.h[0] = prow[4 + hi]; aHi.h[1] = prow[6 + hi];   // K 32..63

        // B fragments: col = l&15; K chunks {kb..kb+7, 16+kb..}
        const v8h* wrow = (const v8h*)W1T[nIdx];
        V16 bLo, bHi;
        bLo.h[0] = wrow[hi];     bLo.h[1] = wrow[2 + hi];
        bHi.h[0] = wrow[4 + hi]; bHi.h[1] = wrow[6 + hi];

        v8f acc = {0.f, 0.f, 0.f, 0.f, 0.f, 0.f, 0.f, 0.f};
        acc = wmma_f16(aLo.v, bLo.v, acc);
        acc = wmma_f16(aHi.v, bHi.v, acc);

        // bias + relu, stage hidden into per-wave LDS tile (f16, K pad 32)
        #pragma unroll
        for (int r = 0; r < 8; ++r) {
            float h = fmaxf(acc[r] + b1n, 0.f);   // D: m = r + hi*8, n = nIdx
            Hs[wave][r + hi * 8][nIdx] = (_Float16)h;
        }
        // same-wave DS ops are in-order: no barrier needed before re-read
        const v8h* hrow = (const v8h*)Hs[wave][nIdx];
        V16 a2;
        a2.h[0] = hrow[hi]; a2.h[1] = hrow[2 + hi];

        const v8h* w2row = (const v8h*)W2T[nIdx];
        V16 bf2;
        bf2.h[0] = w2row[hi]; bf2.h[1] = w2row[2 + hi];

        v8f acc2 = {0.f, 0.f, 0.f, 0.f, 0.f, 0.f, 0.f, 0.f};
        acc2 = wmma_f16(a2.v, bf2.v, acc2);

        if (nIdx < 3) {                            // rgb columns only
            float partial = 0.f;
            #pragma unroll
            for (int r = 0; r < 8; ++r) {
                float z  = acc2[r] + b2n;
                float sg = 1.f / (1.f + __expf(-z));
                partial += wLds[s0 + r + hi * 8] * sg;
            }
            atomicAdd(&accLds[nIdx], partial);     // ds_add_f32
        }
    }

    __syncthreads();
    if (s < 3) out[(size_t)ray * 3 + s] = accLds[s] + bgLds;
}

extern "C" void kernel_launch(void* const* d_in, const int* in_sizes, int n_in,
                              void* d_out, int out_size, void* d_ws, size_t ws_size,
                              hipStream_t stream) {
    (void)n_in; (void)out_size; (void)d_ws; (void)ws_size;
    const float* rays_pts = (const float*)d_in[0];
    const float* viewdirs = (const float*)d_in[1];
    const float* density  = (const float*)d_in[2];
    const float* k0       = (const float*)d_in[3];
    const float* w1       = (const float*)d_in[4];
    const float* b1       = (const float*)d_in[5];
    const float* w2       = (const float*)d_in[6];
    const float* b2       = (const float*)d_in[7];
    float* out = (float*)d_out;

    const int nrays = in_sizes[1] / 3;             // viewdirs = [Nr,3]
    dvgo_fused<<<dim3(nrays), dim3(256), 0, stream>>>(
        rays_pts, viewdirs, density, k0, w1, b1, w2, b2, out);
}
